// CausalSelfAttention_13821204759113
// MI455X (gfx1250) — matrix operand order repaired and checked by reading, not compiled, not run on hardware
//
#include <hip/hip_runtime.h>

typedef __bf16 bf16;
typedef __attribute__((ext_vector_type(8)))  __bf16 bf16x8;
typedef __attribute__((ext_vector_type(16))) __bf16 bf16x16;
typedef __attribute__((ext_vector_type(4)))  float  f32x4;
typedef __attribute__((ext_vector_type(8)))  float  f32x8;

constexpr int kB  = 4;
constexpr int kT  = 1024;
constexpr int kD  = 1024;
constexpr int kH  = 16;
constexpr int kDH = 64;
constexpr int kM  = kB * kT;   // 4096 token rows
constexpr int kN3 = 3 * kD;    // 3072

// CDNA5 async global->LDS copy path (ASYNCcnt-tracked), guarded so the file
// compiles on toolchains that lack the builtins.
#if defined(__has_builtin)
#if __has_builtin(__builtin_amdgcn_global_load_async_to_lds_b128) && \
    __has_builtin(__builtin_amdgcn_s_wait_asynccnt)
#define USE_ASYNC_LDS 1
#endif
#endif
#ifndef USE_ASYNC_LDS
#define USE_ASYNC_LDS 0
#endif

#if USE_ASYNC_LDS
typedef int i32x4v __attribute__((vector_size(16)));                 // <4 x i32>
typedef __attribute__((address_space(1))) i32x4v* gv4i_ptr;          // global
typedef __attribute__((address_space(3))) i32x4v* lv4i_ptr;          // LDS
#endif

__device__ __forceinline__ bf16x16 cat8(bf16x8 lo, bf16x8 hi) {
  return __builtin_shufflevector(lo, hi, 0,1,2,3,4,5,6,7,8,9,10,11,12,13,14,15);
}

__device__ __forceinline__ f32x8 wmma_bf16(bf16x16 a, bf16x16 b, f32x8 c) {
  return __builtin_amdgcn_wmma_f32_16x16x32_bf16(false, a, false, b, (short)0, c, false, false);
}

// A fragment (16x32 bf16). Lane = (h2 = lane>>4, r = lane&15), M = r.
// h2==0 holds K = {0..7, 16..23}; h2==1 holds K = {8..15, 24..31}.
__device__ __forceinline__ bf16x16 afrag_from_f32(const float* row, int h2) {
  f32x4 f0 = *(const f32x4*)(row + 8*h2);
  f32x4 f1 = *(const f32x4*)(row + 8*h2 + 4);
  f32x4 f2 = *(const f32x4*)(row + 16 + 8*h2);
  f32x4 f3 = *(const f32x4*)(row + 16 + 8*h2 + 4);
  bf16x8 lo, hi;
#pragma unroll
  for (int j = 0; j < 4; ++j) {
    lo[j]   = (bf16)f0[j]; lo[j+4] = (bf16)f1[j];
    hi[j]   = (bf16)f2[j]; hi[j+4] = (bf16)f3[j];
  }
  return cat8(lo, hi);
}

__device__ __forceinline__ bf16x16 afrag_from_bf16(const bf16* row, int h2) {
  bf16x8 lo = *(const bf16x8*)(row + 8*h2);
  bf16x8 hi = *(const bf16x8*)(row + 16 + 8*h2);
  return cat8(lo, hi);
}

// B fragment (32x16 bf16): lane n = column, h2==0 holds K = {0..7, 16..23}, h2==1 holds
// K = {8..15, 24..31} (tile stored K-contiguous per column, transposed in LDS).
__device__ __forceinline__ bf16x16 bfrag_from_lds(const bf16* colPtr, int h2) {
  bf16x8 lo = *(const bf16x8*)(colPtr + 8*h2);
  bf16x8 hi = *(const bf16x8*)(colPtr + 16 + 8*h2);
  return cat8(lo, hi);
}

// ---------------- Kernel 1: qkv = x @ Wqkv + bqkv, scatter to q/k/v [B,H,T,DH] bf16
__global__ __launch_bounds__(256) void k_qkv(const float* __restrict__ x,
                                             const float* __restrict__ Wqkv,
                                             const float* __restrict__ bqkv,
                                             bf16* __restrict__ qb,
                                             bf16* __restrict__ kb,
                                             bf16* __restrict__ vb) {
  __shared__ __align__(16) bf16 Bt[128][40];   // W tile, transposed [N][K]
  const int tid = threadIdx.x;
  const int w = tid >> 5, lane = tid & 31;
  const int h2 = lane >> 4, n = lane & 15;
  const int tileM = blockIdx.y * 128, tileN = blockIdx.x * 128;

  f32x8 acc[8] = {};
  for (int k0 = 0; k0 < kD; k0 += 32) {
    __syncthreads();
    if (k0 + 32 < kD)
      __builtin_prefetch(Wqkv + (size_t)(k0 + 32) * kN3 + tileN + (tid & 31) * 4, 0, 1);
#pragma unroll
    for (int it = 0; it < 4; ++it) {
      int idx = tid + it * 256;              // 0..1023
      int kk = idx >> 5;                     // 0..31
      int nn = (idx & 31) * 4;               // 0..124
      f32x4 v = *(const f32x4*)(Wqkv + (size_t)(k0 + kk) * kN3 + tileN + nn);
#pragma unroll
      for (int j = 0; j < 4; ++j) Bt[nn + j][kk] = (bf16)v[j];
    }
    __syncthreads();
    bf16x16 a = afrag_from_f32(x + (size_t)(tileM + 16*w + n) * kD + k0, h2);
    // preload B-fragments in groups of 4 so the WMMAs issue back-to-back
#pragma unroll
    for (int g = 0; g < 2; ++g) {
      bf16x16 bfr[4];
#pragma unroll
      for (int j = 0; j < 4; ++j) bfr[j] = bfrag_from_lds(&Bt[16*(4*g + j) + n][0], h2);
#pragma unroll
      for (int j = 0; j < 4; ++j) acc[4*g + j] = wmma_bf16(a, bfr[j], acc[4*g + j]);
    }
  }

#pragma unroll
  for (int nb = 0; nb < 8; ++nb) {
    const int col  = tileN + 16*nb + n;
    const int seg  = col >> 10;              // 0=q 1=k 2=v (16-wide block never straddles)
    const int d    = col & 1023;
    const int head = d >> 6, dh = d & 63;
    bf16* dst = seg == 0 ? qb : (seg == 1 ? kb : vb);
    const float bias = bqkv[col];
#pragma unroll
    for (int i = 0; i < 8; ++i) {
      const int row = tileM + 16*w + i + 8*h2;
      const int bidx = row >> 10, t = row & 1023;
      dst[(((size_t)(bidx * kH + head)) * kT + t) * kDH + dh] = (bf16)(acc[nb][i] + bias);
    }
  }
}

// ---------------- Kernel 2: flash attention per (b,h,128-row q block)
__global__ __launch_bounds__(256) void k_attn(const bf16* __restrict__ qbuf,
                                              const bf16* __restrict__ kbuf,
                                              const bf16* __restrict__ vbuf,
                                              const int*  __restrict__ mask,
                                              bf16* __restrict__ ab) {
  __shared__ __align__(16) bf16 Ks[32][72];      // K tile, [key][dh]  (B-frag for QK^T)
  __shared__ __align__(16) bf16 Vt[64][40];      // V tile, transposed [dh][key] (B-frag for AV)
  __shared__ __align__(16) bf16 Ps[8][16][40];   // per-wave P transpose scratch

  const int tid = threadIdx.x, w = tid >> 5, lane = tid & 31;
  const int h2 = lane >> 4, n = lane & 15;
  const int bh = blockIdx.y;                 // b*H + h
  const int b  = bh >> 4;
  const int q0 = blockIdx.x * 128;
  const size_t headBase = (size_t)bh * kT * kDH;

  // Pin Q A-fragments (16 rows x 64 dh = two 16x32 frags) in VGPRs
  const int qrow = q0 + 16*w + n;
  bf16x16 aq[2];
#pragma unroll
  for (int c = 0; c < 2; ++c)
    aq[c] = afrag_from_bf16(qbuf + headBase + (size_t)qrow * kDH + 32*c, h2);

  float m_i[8], l_i[8];
  f32x8 o[4] = {};
#pragma unroll
  for (int i = 0; i < 8; ++i) { m_i[i] = -__builtin_inff(); l_i[i] = 0.f; }

  const int nks = blockIdx.x * 4 + 4;        // causal: keys up to q0+127
  for (int ks = 0; ks < nks; ++ks) {
    const int k0 = ks * 32;
    __syncthreads();
    {
      const int key = tid >> 3, seg = tid & 7;   // 32 keys x 8 chunks of 8 halves
      const bf16* gk = kbuf + headBase + (size_t)(k0 + key) * kDH + 8 * seg;
#if USE_ASYNC_LDS
      // ASYNCcnt-tracked DMA of the 4KB K tile straight into LDS
      __builtin_amdgcn_global_load_async_to_lds_b128(
          (gv4i_ptr)gk, (lv4i_ptr)&Ks[key][8 * seg], 0, 0);
#else
      bf16x8 kv = *(const bf16x8*)gk;
      *(bf16x8*)&Ks[key][8 * seg] = kv;
#endif
      // V tile is stored transposed for the AV B-fragments (async copy cannot transpose)
      bf16x8 vv = *(const bf16x8*)(vbuf + headBase + (size_t)(k0 + key) * kDH + 8 * seg);
#pragma unroll
      for (int j = 0; j < 8; ++j) Vt[8 * seg + j][key] = vv[j];
    }
#if USE_ASYNC_LDS
    __builtin_amdgcn_s_wait_asynccnt(0);
#endif
    __syncthreads();

    // S = Q K^T for 16 q-rows x 32 keys (two 16x16 C-frags, K over dh = 2x32)
    f32x8 s0 = {}, s1 = {};
    {
      bf16x16 bk[4];
#pragma unroll
      for (int c = 0; c < 2; ++c) {
        bk[2*c]     = bfrag_from_lds(&Ks[n][32*c], h2);
        bk[2*c + 1] = bfrag_from_lds(&Ks[16 + n][32*c], h2);
      }
#pragma unroll
      for (int c = 0; c < 2; ++c) {
        s0 = wmma_bf16(aq[c], bk[2*c],     s0);
        s1 = wmma_bf16(aq[c], bk[2*c + 1], s1);
      }
    }

    const int key0 = k0 + n, key1 = k0 + 16 + n;
    const bool keep0 = mask[b * kT + key0] != 0;
    const bool keep1 = mask[b * kT + key1] != 0;
    float p0[8], p1[8];
#pragma unroll
    for (int i = 0; i < 8; ++i) {
      const int q = q0 + 16*w + i + 8*h2;
      float v0 = s0[i] * 0.125f;             // 1/sqrt(64)
      float v1 = s1[i] * 0.125f;
      if (key0 > q || !keep0) v0 = -__builtin_inff();
      if (key1 > q || !keep1) v1 = -__builtin_inff();
      // row reductions across the 16-lane half-group owning this C row
      float rm = fmaxf(v0, v1);
#pragma unroll
      for (int mk = 8; mk >= 1; mk >>= 1) rm = fmaxf(rm, __shfl_xor(rm, mk, 32));
      const float mnew = fmaxf(m_i[i], rm);
      const float corr = __expf(m_i[i] - mnew);
      v0 = __expf(v0 - mnew);
      v1 = __expf(v1 - mnew);
      float rs = v0 + v1;
#pragma unroll
      for (int mk = 8; mk >= 1; mk >>= 1) rs += __shfl_xor(rs, mk, 32);
      l_i[i] = l_i[i] * corr + rs;
      m_i[i] = mnew;
#pragma unroll
      for (int nb = 0; nb < 4; ++nb) o[nb][i] *= corr;
      p0[i] = v0; p1[i] = v1;
    }

    // Transpose P (C-frag layout) -> A-frag layout via wave-private LDS scratch
#pragma unroll
    for (int i = 0; i < 8; ++i) {
      Ps[w][i + 8*h2][n]      = (bf16)p0[i];
      Ps[w][i + 8*h2][16 + n] = (bf16)p1[i];
    }
    __syncthreads();
    {
      bf16x16 ap = afrag_from_bf16(&Ps[w][n][0], h2);
      bf16x16 bv[4];
#pragma unroll
      for (int nb = 0; nb < 4; ++nb) bv[nb] = bfrag_from_lds(&Vt[16*nb + n][0], h2);
#pragma unroll
      for (int nb = 0; nb < 4; ++nb) o[nb] = wmma_bf16(ap, bv[nb], o[nb]);
    }
  }

  // normalize and store attention output as [B,T,D] bf16
#pragma unroll
  for (int i = 0; i < 8; ++i) {
    const float inv = 1.f / l_i[i];
    const int q = q0 + 16*w + i + 8*h2;
    const size_t rowBase = ((size_t)(b * kT + q)) * kD + (size_t)(bh & 15) * kDH;
#pragma unroll
    for (int nb = 0; nb < 4; ++nb)
      ab[rowBase + 16*nb + n] = (bf16)(o[nb][i] * inv);
  }
}

// ---------------- Kernel 3: out = attn_out @ Wproj + bproj (f32 out)
__global__ __launch_bounds__(256) void k_proj(const bf16* __restrict__ ain,
                                              const float* __restrict__ Wp,
                                              const float* __restrict__ bp,
                                              float* __restrict__ out) {
  __shared__ __align__(16) bf16 Bt[128][40];
  const int tid = threadIdx.x;
  const int w = tid >> 5, lane = tid & 31;
  const int h2 = lane >> 4, n = lane & 15;
  const int tileM = blockIdx.y * 128, tileN = blockIdx.x * 128;

  f32x8 acc[8] = {};
  for (int k0 = 0; k0 < kD; k0 += 32) {
    __syncthreads();
    if (k0 + 32 < kD)
      __builtin_prefetch(Wp + (size_t)(k0 + 32) * kD + tileN + (tid & 31) * 4, 0, 1);
#pragma unroll
    for (int it = 0; it < 4; ++it) {
      int idx = tid + it * 256;
      int kk = idx >> 5, nn = (idx & 31) * 4;
      f32x4 v = *(const f32x4*)(Wp + (size_t)(k0 + kk) * kD + tileN + nn);
#pragma unroll
      for (int j = 0; j < 4; ++j) Bt[nn + j][kk] = (bf16)v[j];
    }
    __syncthreads();
    bf16x16 a = afrag_from_bf16(ain + (size_t)(tileM + 16*w + n) * kD + k0, h2);
#pragma unroll
    for (int g = 0; g < 2; ++g) {
      bf16x16 bfr[4];
#pragma unroll
      for (int j = 0; j < 4; ++j) bfr[j] = bfrag_from_lds(&Bt[16*(4*g + j) + n][0], h2);
#pragma unroll
      for (int j = 0; j < 4; ++j) acc[4*g + j] = wmma_bf16(a, bfr[j], acc[4*g + j]);
    }
  }

#pragma unroll
  for (int nb = 0; nb < 8; ++nb) {
    const int col = tileN + 16*nb + n;
    const float bias = bp[col];
#pragma unroll
    for (int i = 0; i < 8; ++i) {
      const int row = tileM + 16*w + i + 8*h2;
      out[(size_t)row * kD + col] = acc[nb][i] + bias;
    }
  }
}

extern "C" void kernel_launch(void* const* d_in, const int* in_sizes, int n_in,
                              void* d_out, int out_size, void* d_ws, size_t ws_size,
                              hipStream_t stream) {
  const float* x    = (const float*)d_in[0];
  const float* Wqkv = (const float*)d_in[1];
  const float* bqkv = (const float*)d_in[2];
  const float* Wp   = (const float*)d_in[3];
  const float* bp   = (const float*)d_in[4];
  const int*   mask = (const int*)d_in[5];
  float* out = (float*)d_out;

  const size_t headElems = (size_t)kB * kH * kT * kDH;   // 4M bf16 elems each
  bf16* qb = (bf16*)d_ws;
  bf16* kb = qb + headElems;
  bf16* vb = kb + headElems;
  bf16* ab = vb + headElems;   // total 32 MB of workspace

  k_qkv <<<dim3(kN3 / 128, kM / 128), 256, 0, stream>>>(x, Wqkv, bqkv, qb, kb, vb);
  k_attn<<<dim3(kT / 128, kB * kH),   256, 0, stream>>>(qb, kb, vb, mask, ab);
  k_proj<<<dim3(kD / 128, kM / 128),  256, 0, stream>>>(ab, Wp, bp, out);
}
